// DeltaNetPrefillWrapper_37417755083574
// MI455X (gfx1250) — compile-verified, matrix operand order checked by reading, not run on hardware
//
#include <hip/hip_runtime.h>
#include <hip/hip_bf16.h>

// ---------------------------------------------------------------------------
// Problem constants
// ---------------------------------------------------------------------------
#define Cc   64      // tokens
#define Kc   4       // conv width
#define Hh   4096    // hidden
#define HK   32      // heads
#define DK   128     // head dim (k/q)
#define VE   128     // value dim
#define EPSc 1e-6f

typedef __attribute__((ext_vector_type(16))) __bf16 v16bf;
typedef __attribute__((ext_vector_type(8)))  __bf16 v8bf;
typedef __attribute__((ext_vector_type(8)))  float  v8f;

// ---------------------------------------------------------------------------
// Kernel 1: fp32 [K][64] -> bf16 transposed [64][K] (LDS-tiled, coalesced both sides)
// ---------------------------------------------------------------------------
__global__ __launch_bounds__(256)
void cvt_t_kernel(const float* __restrict__ src, __bf16* __restrict__ dst, int Kdim) {
    __shared__ float t[64][65];
    const int tid = threadIdx.x;
    const int k0  = blockIdx.x * 64;
    #pragma unroll
    for (int i = 0; i < 16; ++i) {
        const int e = tid + i * 256;
        const int kr = e >> 6, c = e & 63;
        t[kr][c] = src[(size_t)(k0 + kr) * Cc + c];
    }
    __syncthreads();
    #pragma unroll
    for (int i = 0; i < 16; ++i) {
        const int e = tid + i * 256;
        const int c = e >> 6, kr = e & 63;
        dst[(size_t)c * Kdim + k0 + kr] = (__bf16)t[kr][c];
    }
}

// ---------------------------------------------------------------------------
// Kernel 2: WMMA bf16 GEMM   C[M][64] = A[M][K](fp32 row-major) * B[K][64]
// B supplied TRANSPOSED as Bt[64][K] bf16. B tile staged into LDS with the
// CDNA5 async global->LDS DMA path (ASYNCcnt), A streamed fp32->bf16 in
// registers. 8 waves; wave => 16x64 strip = 4 WMMA accumulators.
// The async-load LDS address operand is derived from &ldsBt[...] (ptrtoint):
// low 32 bits of the generic address are the LDS byte offset (ISA aperture
// rule), and the escape makes the asm "memory" clobber cover ldsBt so the
// subsequent LDS reads are not folded away.
// ---------------------------------------------------------------------------
__global__ __launch_bounds__(256)
void gemm_wmma_bf16_kernel(const float* __restrict__ A,
                           const float* __restrict__ bias,
                           const __bf16* __restrict__ Bt,
                           float* __restrict__ Cout,
                           int K) {
    __shared__ __bf16 ldsBt[64][72];              // [token][k-chunk], 144B rows (16B aligned)

    const int tid  = threadIdx.x;
    const int wave = tid >> 5;
    const int lane = tid & 31;
    const int r    = lane & 15;
    const int g    = lane >> 4;
    const int rowBase = blockIdx.x * 128 + wave * 16;

    v8f acc[4];
    #pragma unroll
    for (int t = 0; t < 4; ++t) acc[t] = (v8f){0.f,0.f,0.f,0.f,0.f,0.f,0.f,0.f};

    const float* arow = A + (size_t)(rowBase + r) * K;

    // staging geometry: 64 rows x 64 k of bf16 = 8KB; thread moves 32B
    const int srow = tid >> 2, sseg = tid & 3;
    const unsigned ldsoff = (unsigned)(size_t)(&ldsBt[srow][sseg * 16]);

    for (int kk = 0; kk < K; kk += 64) {
        __syncthreads();                          // all waves done reading prev tile
        {   // async DMA: Bt[srow][kk + sseg*16 .. +15] -> LDS (2 x 16B)
            unsigned long long gaddr =
                (unsigned long long)(uintptr_t)&Bt[(size_t)srow * K + kk + sseg * 16];
            asm volatile(
                "global_load_async_to_lds_b128 %0, %1, off\n\t"
                "global_load_async_to_lds_b128 %0, %1, off offset:16\n\t"
                "s_wait_asynccnt 0"
                :: "v"(ldsoff), "v"(gaddr) : "memory");
        }
        __syncthreads();                          // tile visible to all waves

        #pragma unroll
        for (int ks = 0; ks < 2; ++ks) {
            // ---- A fragment (16x32 bf16, ISA layout): half-wave g==0 holds
            //      K {0..7,16..23}, g==1 holds K {8..15,24..31} ----
            const float* ap = arow + kk + ks * 32;
            __builtin_prefetch(ap + 128, 0, 2);   // global_prefetch_b8 (two tiles ahead)
            const float4* p0 = (const float4*)(ap + g * 8);
            const float4* p1 = (const float4*)(ap + 16 + g * 8);
            float4 a0 = p0[0], a1 = p0[1], a2 = p1[0], a3 = p1[1];
            v16bf afrag;
            afrag[0]=(__bf16)a0.x; afrag[1]=(__bf16)a0.y; afrag[2]=(__bf16)a0.z; afrag[3]=(__bf16)a0.w;
            afrag[4]=(__bf16)a1.x; afrag[5]=(__bf16)a1.y; afrag[6]=(__bf16)a1.z; afrag[7]=(__bf16)a1.w;
            afrag[8]=(__bf16)a2.x; afrag[9]=(__bf16)a2.y; afrag[10]=(__bf16)a2.z; afrag[11]=(__bf16)a2.w;
            afrag[12]=(__bf16)a3.x; afrag[13]=(__bf16)a3.y; afrag[14]=(__bf16)a3.z; afrag[15]=(__bf16)a3.w;

            // ---- preload all 4 B fragments (8 x ds_load_b128, one clause) ----
            v16bf bfr[4];
            #pragma unroll
            for (int nt = 0; nt < 4; ++nt) {
                const int n = nt * 16 + r;
                v8bf blo = *(const v8bf*)&ldsBt[n][ks * 32 + g * 8];
                v8bf bhi = *(const v8bf*)&ldsBt[n][ks * 32 + 16 + g * 8];
                bfr[nt] = __builtin_shufflevector(blo, bhi,
                    0,1,2,3,4,5,6,7,8,9,10,11,12,13,14,15);
            }
            // ---- 4 back-to-back WMMAs on independent accumulators ----
            #pragma unroll
            for (int nt = 0; nt < 4; ++nt)
                acc[nt] = __builtin_amdgcn_wmma_f32_16x16x32_bf16(
                    false, afrag, false, bfr[nt], (short)0, acc[nt], false, false);
        }
    }

    // ---- epilogue: D layout M = j + 8*g, N = r ----
    #pragma unroll
    for (int j = 0; j < 8; ++j) {
        const int row = rowBase + j + 8 * g;
        const float bv = bias[row];
        #pragma unroll
        for (int nt = 0; nt < 4; ++nt)
            Cout[(size_t)row * Cc + nt * 16 + r] = acc[nt][j] + bv;
    }
}

// ---------------------------------------------------------------------------
// Kernel 3: alpha/beta = sigmoid(Wa x + ba), sigmoid(Wb x + bb)
// ---------------------------------------------------------------------------
__global__ void ab_kernel(const float* __restrict__ Wa, const float* __restrict__ ba,
                          const float* __restrict__ Wb, const float* __restrict__ bb,
                          const float* __restrict__ x,
                          float* __restrict__ alpha, float* __restrict__ beta) {
    const int h = blockIdx.x;
    const int c = threadIdx.x;
    const float* wa = Wa + (size_t)h * Hh;
    const float* wb = Wb + (size_t)h * Hh;
    float aacc = 0.f, bacc = 0.f;
    for (int k = 0; k < Hh; ++k) {
        const float xv = x[(size_t)k * Cc + c];
        aacc += wa[k] * xv;
        bacc += wb[k] * xv;
    }
    aacc += ba[h]; bacc += bb[h];
    alpha[h * Cc + c] = 1.f / (1.f + __expf(-aacc));
    beta [h * Cc + c] = 1.f / (1.f + __expf(-bacc));
}

// ---------------------------------------------------------------------------
// Kernel 4: depthwise causal conv(K=4) + SiLU (+ per-head L2 norm for q,k)
// grid.x = head*token (2048), grid.y = {0:q,1:k,2:v}, 128 threads = d
// ---------------------------------------------------------------------------
__global__ __launch_bounds__(128)
void conv_norm_kernel(const float* __restrict__ qlin, const float* __restrict__ klin,
                      const float* __restrict__ vlin,
                      const float* __restrict__ qw, const float* __restrict__ qb,
                      const float* __restrict__ kw, const float* __restrict__ kb,
                      const float* __restrict__ vw, const float* __restrict__ vb,
                      float* __restrict__ qn, float* __restrict__ kn,
                      float* __restrict__ vp) {
    const int h = blockIdx.x >> 6;
    const int c = blockIdx.x & 63;
    const int d = threadIdx.x;
    const int o = h * DK + d;
    const int sel = blockIdx.y;

    const float* src = (sel == 0) ? qlin : (sel == 1) ? klin : vlin;
    const float* w   = (sel == 0) ? qw   : (sel == 1) ? kw   : vw;
    const float* bb_ = (sel == 0) ? qb   : (sel == 1) ? kb   : vb;
    float*       dst = (sel == 0) ? qn   : (sel == 1) ? kn   : vp;

    float y = bb_[o];
    #pragma unroll
    for (int t = 0; t < Kc; ++t) {
        const int ci = c - (Kc - 1) + t;
        if (ci >= 0) y += w[o * Kc + t] * src[(size_t)o * Cc + ci];
    }
    y = y / (1.f + __expf(-y));                   // SiLU

    if (sel < 2) {                                // L2 normalize over d
        __shared__ float red[128];
        red[d] = y * y;
        __syncthreads();
        for (int off = 64; off > 0; off >>= 1) {
            if (d < off) red[d] += red[d + off];
            __syncthreads();
        }
        y *= rsqrtf(red[0] + EPSc);
    }
    dst[((size_t)h * Cc + c) * DK + d] = y;
}

// ---------------------------------------------------------------------------
// Kernel 5: DeltaNet recurrence. One block per head, thread e owns state[:,e]
// (128 f32 in VGPRs); k/q broadcast via LDS => all dots per-thread.
// Emits core output TRANSPOSED bf16 core_t[c][v=h*128+e] for the Wo GEMM.
// ---------------------------------------------------------------------------
__global__ __launch_bounds__(128)
void scan_kernel(const float* __restrict__ qn, const float* __restrict__ kn,
                 const float* __restrict__ vp,
                 const float* __restrict__ alpha, const float* __restrict__ beta,
                 const float* __restrict__ state0,
                 __bf16* __restrict__ core_t) {
    const int h = blockIdx.x;
    const int e = threadIdx.x;

    float s[DK];
    #pragma unroll
    for (int d = 0; d < DK; ++d)
        s[d] = state0[((size_t)h * DK + d) * VE + e];

    __shared__ float lk[DK], lq[DK];

    for (int c = 0; c < Cc; ++c) {
        const size_t base = ((size_t)h * Cc + c) * DK;
        lk[e] = kn[base + e];
        lq[e] = qn[base + e];
        __syncthreads();
        const float at = alpha[h * Cc + c];
        const float bt = beta [h * Cc + c];
        const float ve = vp[base + e];

        float retr = 0.f;
        #pragma unroll
        for (int d = 0; d < DK; ++d) { s[d] *= at; retr += s[d] * lk[d]; }

        const float err = (ve - retr) * bt;

        float ot = 0.f;
        #pragma unroll
        for (int d = 0; d < DK; ++d) { s[d] += lk[d] * err; ot += s[d] * lq[d]; }

        core_t[(size_t)c * (HK * VE) + h * VE + e] = (__bf16)ot;
        __syncthreads();
    }
}

// ---------------------------------------------------------------------------
// Host launcher
// ---------------------------------------------------------------------------
extern "C" void kernel_launch(void* const* d_in, const int* in_sizes, int n_in,
                              void* d_out, int out_size, void* d_ws, size_t ws_size,
                              hipStream_t stream) {
    const float* hidden = (const float*)d_in[0];
    const float* Wq = (const float*)d_in[1];   const float* bq = (const float*)d_in[2];
    const float* Wk = (const float*)d_in[3];   const float* bk = (const float*)d_in[4];
    const float* Wv = (const float*)d_in[5];   const float* bv = (const float*)d_in[6];
    const float* Wo = (const float*)d_in[7];   const float* bo = (const float*)d_in[8];
    const float* qw = (const float*)d_in[9];   const float* qcb = (const float*)d_in[10];
    const float* kw = (const float*)d_in[11];  const float* kcb = (const float*)d_in[12];
    const float* vw = (const float*)d_in[13];  const float* vcb = (const float*)d_in[14];
    const float* Wa = (const float*)d_in[15];  const float* ba = (const float*)d_in[16];
    const float* Wb = (const float*)d_in[17];  const float* bb = (const float*)d_in[18];
    const float* state0 = (const float*)d_in[19];
    float* out = (float*)d_out;

    char* ws = (char*)d_ws;
    const size_t sz_xb  = (size_t)Hh * Cc * sizeof(__bf16);      // 512 KB
    const size_t sz_lin = (size_t)Hh * Cc * sizeof(float);       // 1 MB
    const size_t sz_hd  = (size_t)HK * Cc * DK * sizeof(float);  // 1 MB
    const size_t sz_ab  = (size_t)HK * Cc * sizeof(float);

    size_t off = 0;
    __bf16* x_bf_t = (__bf16*)(ws + off); off += sz_xb;          // [64][4096]
    float*  q_lin  = (float*) (ws + off); off += sz_lin;
    float*  k_lin  = (float*) (ws + off); off += sz_lin;
    float*  v_lin  = (float*) (ws + off); off += sz_lin;
    float*  q_n    = (float*) (ws + off); off += sz_hd;
    float*  k_n    = (float*) (ws + off); off += sz_hd;
    float*  v_p    = (float*) (ws + off); off += sz_hd;
    float*  al     = (float*) (ws + off); off += sz_ab;
    float*  be     = (float*) (ws + off); off += sz_ab;
    __bf16* core_t = (__bf16*)(ws + off); off += sz_xb;          // [64][4096]

    // 1) transpose-convert x -> bf16 [c][k] (reused from L2 by all 3 GEMMs)
    cvt_t_kernel<<<Hh / 64, 256, 0, stream>>>(hidden, x_bf_t, Hh);

    // 2) Q/K/V projections: M=4096, N=64, K=4096 WMMA GEMMs
    gemm_wmma_bf16_kernel<<<Hh / 128, 256, 0, stream>>>(Wq, bq, x_bf_t, q_lin, Hh);
    gemm_wmma_bf16_kernel<<<Hh / 128, 256, 0, stream>>>(Wk, bk, x_bf_t, k_lin, Hh);
    gemm_wmma_bf16_kernel<<<Hh / 128, 256, 0, stream>>>(Wv, bv, x_bf_t, v_lin, Hh);

    // 3) alpha / beta gates
    ab_kernel<<<HK, Cc, 0, stream>>>(Wa, ba, Wb, bb, hidden, al, be);

    // 4) causal depthwise conv + SiLU (+ L2 norm for q,k), relayout to [h][c][d]
    conv_norm_kernel<<<dim3(HK * Cc, 3), DK, 0, stream>>>(
        q_lin, k_lin, v_lin, qw, qcb, kw, kcb, vw, vcb, q_n, k_n, v_p);

    // 5) sequential DeltaNet scan, one WGP per head, state in VGPRs
    scan_kernel<<<HK, DK, 0, stream>>>(q_n, k_n, v_p, al, be, state0, core_t);

    // 6) output projection Wo @ core + bo -> d_out  ( (1,H,1,C) == [o][c] )
    gemm_wmma_bf16_kernel<<<Hh / 128, 256, 0, stream>>>(Wo, bo, core_t, out, Hh);
}
